// Wavelet_25400436589286
// MI455X (gfx1250) — matrix operand order repaired and checked by reading, not compiled, not run on hardware
//
#include <hip/hip_runtime.h>
#include <math.h>

#ifndef M_PI
#define M_PI 3.14159265358979323846
#endif

typedef __attribute__((ext_vector_type(16))) _Float16 v16h;
typedef __attribute__((ext_vector_type(8)))  _Float16 v8h;
typedef __attribute__((ext_vector_type(8)))  float    v8f;
typedef __attribute__((ext_vector_type(2)))  float    v2f;

#define NB      8
#define T_TOT   672
#define NCH     32
#define FF      64
#define NFREQ   337
#define NSCALES 8
#define PRED    336
#define SMAXP   1344
#define SMAXW   (NSCALES * T_TOT)   // 5376
#define KWIN    11
#define PADW    5
#define PC10    0.0009765625f       // 0.5^10

struct ScaleParams {
    int    L[8];      // filter length (taps), diff filter has L+1 taps
    int    off[8];    // crop offset
    float  nsq[8];    // -sqrt(scale)
    double sc[8];     // scale
};

// ---------------------------------------------------------------- utilities
__device__ inline float gelu_exact(float v) {
    return 0.5f * v * (1.f + erff(v * 0.70710678118654752f));
}

// ---------------------------------------------------------------- tables
__global__ void k_twiddle(float* twc, float* tws) {
    int j = blockIdx.x * blockDim.x + threadIdx.x;
    if (j < T_TOT) {
        double a = (2.0 * M_PI * (double)j) / (double)T_TOT;
        twc[j] = (float)cos(a);
        tws[j] = (float)sin(a);
    }
}

__global__ void k_intpsi(float* intpsi) {
    if (threadIdx.x == 0 && blockIdx.x == 0) {
        const double stepx = 16.0 / 1023.0;
        double acc = 0.0;
        for (int g = 0; g < 1024; ++g) {
            double xv = -8.0 + (double)g * stepx;
            acc += exp(-xv * xv * 0.5) * cos(5.0 * xv);
            intpsi[g] = (float)(acc * stepx);
        }
    }
}

__global__ void k_dfilt(const float* __restrict__ intpsi, float* __restrict__ dfilt,
                        ScaleParams sp) {
    int m = blockIdx.x * blockDim.x + threadIdx.x;
    int s = blockIdx.y;
    if (m >= 1032) return;
    int L = sp.L[s];
    double sstep = sp.sc[s] * (16.0 / 1023.0);
    float v = 0.f;
    if (m <= L) {
        float fm = 0.f, fm1 = 0.f;
        if (m <= L - 1) { int jj = (int)((double)(L - 1 - m) / sstep); fm  = intpsi[jj]; }
        if (m >= 1)     { int jj = (int)((double)(L - m)     / sstep); fm1 = intpsi[jj]; }
        v = fm - fm1;
    }
    dfilt[s * 1032 + m] = v;
}

// ---------------------------------------------------------------- weights
__global__ void k_meanbias(const float* __restrict__ b1, const float* __restrict__ b2,
                           float* bm1, float* bm2) {
    int t = threadIdx.x;
    if (t < 64) {
        float s = 0.f;
        for (int j = 0; j < 6; ++j) s += b1[j * 64 + t];
        bm1[t] = s * (1.f / 6.f);
    } else if (t < 96) {
        int c = t - 64;
        float s = 0.f;
        for (int j = 0; j < 6; ++j) s += b2[j * 32 + c];
        bm2[c] = s * (1.f / 6.f);
    }
}

// conv1 weights (64 out, 32 in): B-tile layout: lane -> (N=nt*16+(lane&15), Khalf=lane>>4)
__global__ void k_swz1(const float* __restrict__ w1, _Float16* __restrict__ bswz) {
    int idx = blockIdx.x * blockDim.x + threadIdx.x;
    if (idx >= 121 * 4 * 512) return;
    int e = idx & 15, lane = (idx >> 4) & 31, nt = (idx >> 9) & 3, o = idx >> 11;
    int ky = o / KWIN, kx = o % KWIN;
    int N = nt * 16 + (lane & 15);
    int K = (lane >> 4) * 16 + e;
    float s = 0.f;
    for (int j = 0; j < 6; ++j)
        s += w1[((((size_t)j * 64 + N) * 32 + K) * KWIN + ky) * KWIN + kx];
    bswz[idx] = (_Float16)(s * (1.f / 6.f));
}

// conv2 weights (32 out, 64 in): two K-chunks of 32
__global__ void k_swz2(const float* __restrict__ w2, _Float16* __restrict__ bswz) {
    int idx = blockIdx.x * blockDim.x + threadIdx.x;
    if (idx >= 121 * 2 * 2 * 512) return;
    int e = idx & 15, lane = (idx >> 4) & 31, nt = (idx >> 9) & 1, kc = (idx >> 10) & 1,
        o = idx >> 11;
    int ky = o / KWIN, kx = o % KWIN;
    int N = nt * 16 + (lane & 15);
    int K = kc * 32 + (lane >> 4) * 16 + e;
    float s = 0.f;
    for (int j = 0; j < 6; ++j)
        s += w2[((((size_t)j * 32 + N) * 64 + K) * KWIN + ky) * KWIN + kx];
    bswz[idx] = (_Float16)(s * (1.f / 6.f));
}

// ---------------------------------------------------------------- DFT amp + top-k
__global__ void k_dft_amp(const float* __restrict__ x, const float* __restrict__ twc,
                          const float* __restrict__ tws, float* __restrict__ ampn) {
    int k = blockIdx.x, b = blockIdx.y, n = threadIdx.x;
    float re = 0.f, im = 0.f;
    for (int t = 0; t < T_TOT; ++t) {
        int j = (k * t) % T_TOT;
        float xv = x[((size_t)(b * T_TOT + t)) * NCH + n];
        re += xv * twc[j];
        im -= xv * tws[j];
    }
    float a = sqrtf(re * re + im * im);
    for (int off = 16; off > 0; off >>= 1) a += __shfl_down(a, off, 32);
    if (n == 0) ampn[b * NFREQ + k] = a * (1.f / NCH);
}

__global__ void k_topk(const float* __restrict__ ampn, int* __restrict__ topidx,
                       float* __restrict__ pw, int* __restrict__ jobs) {
    __shared__ float sf[NFREQ];
    __shared__ int   sidx[3];
    int tid = threadIdx.x;
    if (tid < NFREQ) {
        float s = 0.f;
        for (int b = 0; b < NB; ++b) s += ampn[b * NFREQ + tid];
        sf[tid] = (tid == 0) ? 0.f : s * (1.f / NB);
    }
    __syncthreads();
    if (tid == 0) {
        for (int k = 0; k < 3; ++k) {
            int best = 1; float bv = -1.f;
            for (int f = 1; f < NFREQ; ++f)
                if (sf[f] > bv) { bv = sf[f]; best = f; }
            sidx[k] = best; topidx[k] = best; sf[best] = -2.f;
            int p = T_TOT / best;
            int length = (T_TOT % p == 0) ? T_TOT : (T_TOT / p + 1) * p;
            jobs[k * 4 + 0] = length / p;
            jobs[k * 4 + 1] = p;
            jobs[k * 4 + 2] = length;
            jobs[k * 4 + 3] = 0;
        }
        jobs[12] = NSCALES; jobs[13] = T_TOT; jobs[14] = SMAXW; jobs[15] = 0;
    }
    __syncthreads();
    if (tid < NB) {
        float w[3], mx = -1e30f;
        for (int k = 0; k < 3; ++k) { w[k] = ampn[tid * NFREQ + sidx[k]]; mx = fmaxf(mx, w[k]); }
        float se = 0.f;
        for (int k = 0; k < 3; ++k) { w[k] = expf(w[k] - mx); se += w[k]; }
        for (int k = 0; k < 3; ++k) pw[tid * 3 + k] = w[k] / se;
    }
}

// ---------------------------------------------------------------- inputs for convs
__global__ void k_fill_imp(const float* __restrict__ x, _Float16* __restrict__ imp) {
    int idx = blockIdx.x * blockDim.x + threadIdx.x;
    if (idx >= NB * SMAXP * NCH) return;
    int n = idx & 31, pos = (idx >> 5) % SMAXP, b = idx / (SMAXP * NCH);
    imp[idx] = (pos < T_TOT) ? (_Float16)x[((size_t)(b * T_TOT + pos)) * NCH + n]
                             : (_Float16)0.f;
}

__global__ void k_cwt(const float* __restrict__ x, const float* __restrict__ dfilt,
                      _Float16* __restrict__ cwin, ScaleParams sp) {
    int idx = blockIdx.x * blockDim.x + threadIdx.x;
    if (idx >= NB * NSCALES * T_TOT * NCH) return;
    int n = idx & 31; int rest = idx >> 5;
    int tt = rest % T_TOT; rest /= T_TOT;
    int s = rest & 7; int b = rest >> 3;
    int L = sp.L[s], off = sp.off[s];
    const float* df = dfilt + s * 1032;
    int base = tt + off + 1;
    int m0 = base - (T_TOT - 1); if (m0 < 0) m0 = 0;
    int m1 = base;               if (m1 > L) m1 = L;
    float acc = 0.f;
    for (int m = m0; m <= m1; ++m) {
        int ri = base - m;
        acc += df[m] * x[((size_t)(b * T_TOT + ri)) * NCH + n];
    }
    cwin[((size_t)(b * SMAXW + s * T_TOT + tt)) * NCH + n] = (_Float16)(sp.nsq[s] * acc);
}

// ---------------------------------------------------------------- WMMA conv (implicit GEMM)
__global__ void __launch_bounds__(32) k_conv(
    const _Float16* __restrict__ im, int im_stride, int cin,
    const _Float16* __restrict__ bswz, const float* __restrict__ bias, int cout,
    const int* __restrict__ jobs, int jobidx, int act,
    _Float16* __restrict__ out_h, float* __restrict__ out_f)
{
    const int H = jobs[jobidx * 4 + 0];
    const int W = jobs[jobidx * 4 + 1];
    const int S = jobs[jobidx * 4 + 2];
    const int tm = blockIdx.x;
    if (tm * 16 >= S) return;
    const int nt = blockIdx.y, b = blockIdx.z;
    const int lane = threadIdx.x;
    const int r = lane & 15, hi = lane >> 4;
    const int kchunks = cin >> 5, ntiles = cout >> 4;
    const int m = tm * 16 + r;
    const int h = m / W, w = m - h * W;
    v8f acc = {};
    for (int o = 0; o < KWIN * KWIN; ++o) {
        const int ky = o / KWIN - PADW, kx = o % KWIN - PADW;
        const int ih = h + ky, iw = w + kx;
        const bool valid = (m < S) && ((unsigned)ih < (unsigned)H) && ((unsigned)iw < (unsigned)W);
        const _Float16* ap  = im + (size_t)(b * im_stride + ih * W + iw) * cin;
        const _Float16* bp0 = bswz + ((size_t)(o * kchunks) * ntiles + nt) * 512 + (size_t)lane * 16;
        __builtin_prefetch(bp0 + (size_t)kchunks * ntiles * 512, 0, 1);
        for (int kc = 0; kc < kchunks; ++kc) {
            v16h a = {};
            if (valid) {
                a = *(const v16h*)(ap + kc * 32 + 16 * hi);
            }
            const v16h bv = *(const v16h*)(bp0 + (size_t)kc * ntiles * 512);
            acc = __builtin_amdgcn_wmma_f32_16x16x32_f16(false, a, false, bv,
                                                         (short)0, acc, false, false);
        }
    }
    const int nidx = nt * 16 + r;
    const float bsv = bias[nidx];
    for (int e = 0; e < 8; ++e) {
        const int pos = tm * 16 + 8 * hi + e;
        if (pos < S) {
            float v = acc[e] + bsv;
            size_t oidx = (size_t)(b * im_stride + pos) * cout + nidx;
            if (act) out_h[oidx] = (_Float16)gelu_exact(v);
            else     out_f[oidx] = v;
        }
    }
}

// ---------------------------------------------------------------- combine branches
__global__ void k_combine(const float* __restrict__ wout,
                          const float* __restrict__ p0, const float* __restrict__ p1,
                          const float* __restrict__ p2, const float* __restrict__ pw,
                          const float* __restrict__ wscale, const float* __restrict__ bscale,
                          const float* __restrict__ x, float* __restrict__ acc) {
    int idx = blockIdx.x * blockDim.x + threadIdx.x;
    if (idx >= NB * T_TOT * NCH) return;
    int n = idx & 31, t = (idx >> 5) % T_TOT, b = idx / (T_TOT * NCH);
    float wr = bscale[n];
    for (int s = 0; s < NSCALES; ++s)
        wr += wout[((size_t)(b * SMAXW + s * T_TOT + t)) * NCH + n] * wscale[n * NSCALES + s];
    size_t pidx = ((size_t)(b * SMAXP + t)) * NCH + n;
    float pr = pw[b * 3 + 0] * p0[pidx] + pw[b * 3 + 1] * p1[pidx] + pw[b * 3 + 2] * p2[pidx];
    acc[idx] = x[idx] + (1.f - PC10) * wr + PC10 * pr;
}

// ---------------------------------------------------------------- transposes / DCT
__global__ void k_tr_btn_bnt(const float* __restrict__ src, float* __restrict__ dst) {
    int idx = blockIdx.x * blockDim.x + threadIdx.x;
    if (idx >= NB * T_TOT * NCH) return;
    int n = idx & 31, t = (idx >> 5) % T_TOT, b = idx / (T_TOT * NCH);
    dst[((size_t)(b * NCH + n)) * T_TOT + t] = src[idx];
}

__global__ void k_dctmat(float* __restrict__ Cm) {
    int idx = blockIdx.x * blockDim.x + threadIdx.x;
    if (idx >= T_TOT * T_TOT) return;
    int l = idx / T_TOT, k = idx % T_TOT;
    Cm[idx] = (float)(2.0 * cos(M_PI * (double)k * (2.0 * l + 1.0) / (2.0 * T_TOT)));
}

// ---------------------------------------------------------------- fp32 WMMA GEMM
__global__ void __launch_bounds__(32) k_gemm_f32(
    const float* __restrict__ A, int lda,
    const float* __restrict__ Bm, int ldb, int btrans,
    const float* __restrict__ bias, float* __restrict__ Cout,
    int M, int N, int K, int mode)
{
    int tm = blockIdx.x, tn = blockIdx.y;
    int lane = threadIdx.x, r = lane & 15, hi = lane >> 4;
    int arow = tm * 16 + r;
    int ncol = tn * 16 + r;
    v8f acc = {};
    for (int kk = 0; kk < K; kk += 4) {
        v2f av, bv;
        av[0] = A[(size_t)arow * lda + kk + 2 * hi + 0];
        av[1] = A[(size_t)arow * lda + kk + 2 * hi + 1];
        if (btrans) {
            bv[0] = Bm[(size_t)ncol * ldb + kk + 2 * hi + 0];
            bv[1] = Bm[(size_t)ncol * ldb + kk + 2 * hi + 1];
        } else {
            bv[0] = Bm[(size_t)(kk + 2 * hi + 0) * ldb + ncol];
            bv[1] = Bm[(size_t)(kk + 2 * hi + 1) * ldb + ncol];
        }
        acc = __builtin_amdgcn_wmma_f32_16x16x4_f32(false, av, false, bv,
                                                    (short)0, acc, false, false);
    }
    for (int e = 0; e < 8; ++e) {
        int mrow = tm * 16 + 8 * hi + e;
        int nc = tn * 16 + r;
        if (mrow < M && nc < N) {
            float v = acc[e];
            if (bias) v += bias[nc];
            if (mode == 0) {
                Cout[(size_t)mrow * N + nc] = v;
            } else {
                int b = mrow >> 5, n = mrow & 31;
                Cout[((size_t)b * PRED + nc) * NCH + n] = v;
            }
        }
    }
}

// ---------------------------------------------------------------- LN stats + FC gate
__global__ void k_rowstats(const float* __restrict__ f, float* __restrict__ stats) {
    int r = threadIdx.x;
    if (r >= NB * NCH) return;
    float s = 0.f;
    for (int l = 0; l < T_TOT; ++l) s += f[(size_t)r * T_TOT + l];
    float mu = s / T_TOT;
    float v = 0.f;
    for (int l = 0; l < T_TOT; ++l) { float d = f[(size_t)r * T_TOT + l] - mu; v += d * d; }
    v /= T_TOT;
    stats[r * 2 + 0] = mu;
    stats[r * 2 + 1] = rsqrtf(v + 1e-6f);
}

__global__ void k_fcblock(const float* __restrict__ f, const float* __restrict__ stats,
                          const float* __restrict__ xcl, const float* __restrict__ lnw,
                          const float* __restrict__ lnb, const float* __restrict__ w1,
                          const float* __restrict__ w2, const float* __restrict__ x,
                          float* __restrict__ out2) {
    int idx = blockIdx.x * blockDim.x + threadIdx.x;
    if (idx >= NB * T_TOT) return;
    int l = idx % T_TOT, b = idx / T_TOT;
    float lw = lnw[l], lb = lnb[l];
    float lnv[NCH];
    for (int c = 0; c < NCH; ++c) {
        int row = b * NCH + c;
        lnv[c] = (f[(size_t)row * T_TOT + l] - stats[row * 2]) * stats[row * 2 + 1] * lw + lb;
    }
    float hb[FF];
    for (int d = 0; d < FF; ++d) {
        float s = 0.f;
        for (int c = 0; c < NCH; ++c) s += lnv[c] * w1[d * NCH + c];
        hb[d] = fmaxf(s, 0.f);
    }
    for (int c = 0; c < NCH; ++c) {
        float s = 0.f;
        for (int d = 0; d < FF; ++d) s += hb[d] * w2[c * FF + d];
        float lr = 1.f / (1.f + expf(-s));
        out2[((size_t)(b * T_TOT + l)) * NCH + c] =
            xcl[((size_t)(b * NCH + c)) * T_TOT + l] * lr +
            x[((size_t)(b * T_TOT + l)) * NCH + c];
    }
}

// ================================================================ launcher
extern "C" void kernel_launch(void* const* d_in, const int* in_sizes, int n_in,
                              void* d_out, int out_size, void* d_ws, size_t ws_size,
                              hipStream_t stream) {
    const float* x       = (const float*)d_in[0];
    const float* w_inc1  = (const float*)d_in[1];
    const float* b_inc1  = (const float*)d_in[2];
    const float* w_inc2  = (const float*)d_in[3];
    const float* b_inc2  = (const float*)d_in[4];
    const float* w_scale = (const float*)d_in[5];
    const float* b_scale = (const float*)d_in[6];
    const float* ln_w    = (const float*)d_in[7];
    const float* ln_b    = (const float*)d_in[8];
    const float* w_fc1   = (const float*)d_in[9];
    const float* w_fc2   = (const float*)d_in[10];
    const float* w_proj  = (const float*)d_in[11];
    const float* b_proj  = (const float*)d_in[12];
    (void)in_sizes; (void)n_in; (void)out_size; (void)ws_size;

    // static (input-independent) wavelet filter geometry, matching numpy
    ScaleParams sp;
    const double step = 16.0 / 1023.0;
    for (int i = 0; i < 8; ++i) {
        double e = (i == 7) ? 4.0 : (-1.0 + 5.0 * (double)i / 7.0);
        double s = pow(2.0, e);
        int cnt = (int)ceil(16.0 * s + 1.0);
        int L = 0;
        for (int m = 0; m < cnt; ++m) {
            long j = (long)((double)m / (s * step));
            if (j < 1024) ++L; else break;
        }
        sp.L[i]   = L;
        sp.off[i] = (L - 2) / 2;
        sp.nsq[i] = (float)(-sqrt(s));
        sp.sc[i]  = s;
    }

    // workspace carve-up
    uint8_t* wsb = (uint8_t*)d_ws;
    size_t off = 0;
    auto alloc = [&](size_t bytes) -> void* {
        void* p = wsb + off;
        off = (off + bytes + 255) & ~(size_t)255;
        return p;
    };
    float*    twc    = (float*)alloc(T_TOT * 4);
    float*    tws    = (float*)alloc(T_TOT * 4);
    float*    intpsi = (float*)alloc(1024 * 4);
    float*    dfilt  = (float*)alloc(8 * 1032 * 4);
    float*    ampn   = (float*)alloc(NB * NFREQ * 4);
    int*      topidx = (int*)alloc(16 * 4);
    float*    pw     = (float*)alloc(32 * 4);
    int*      jobs   = (int*)alloc(16 * 4);
    float*    bm1    = (float*)alloc(64 * 4);
    float*    bm2    = (float*)alloc(32 * 4);
    _Float16* bswz1  = (_Float16*)alloc((size_t)121 * 4 * 512 * 2);
    _Float16* bswz2  = (_Float16*)alloc((size_t)121 * 2 * 2 * 512 * 2);
    _Float16* imp    = (_Float16*)alloc((size_t)NB * SMAXP * NCH * 2);
    _Float16* cwin   = (_Float16*)alloc((size_t)NB * SMAXW * NCH * 2);
    _Float16* mid    = (_Float16*)alloc((size_t)NB * SMAXW * FF * 2);
    float*    pout0  = (float*)alloc((size_t)NB * SMAXP * NCH * 4);
    float*    pout1  = (float*)alloc((size_t)NB * SMAXP * NCH * 4);
    float*    pout2  = (float*)alloc((size_t)NB * SMAXP * NCH * 4);
    float*    wout   = (float*)alloc((size_t)NB * SMAXW * NCH * 4);
    float*    accb   = (float*)alloc((size_t)NB * T_TOT * NCH * 4);
    float*    xcl    = (float*)alloc((size_t)NB * NCH * T_TOT * 4);
    float*    Cm     = (float*)alloc((size_t)T_TOT * T_TOT * 4);
    float*    fbuf   = (float*)alloc((size_t)NB * NCH * T_TOT * 4);
    float*    stats  = (float*)alloc(512 * 4);
    float*    out2   = (float*)alloc((size_t)NB * T_TOT * NCH * 4);
    float*    aproj  = (float*)alloc((size_t)NB * NCH * T_TOT * 4);
    float*    pouts[3] = { pout0, pout1, pout2 };

    // tables
    k_twiddle<<<3, 256, 0, stream>>>(twc, tws);
    k_intpsi<<<1, 1, 0, stream>>>(intpsi);
    k_dfilt<<<dim3(5, 8), 256, 0, stream>>>(intpsi, dfilt, sp);
    k_meanbias<<<1, 96, 0, stream>>>(b_inc1, b_inc2, bm1, bm2);
    k_swz1<<<(121 * 4 * 512 + 255) / 256, 256, 0, stream>>>(w_inc1, bswz1);
    k_swz2<<<(121 * 2 * 2 * 512 + 255) / 256, 256, 0, stream>>>(w_inc2, bswz2);

    // spectrum, top-k periods, softmax weights, job geometry
    k_dft_amp<<<dim3(NFREQ, NB), 32, 0, stream>>>(x, twc, tws, ampn);
    k_topk<<<1, 352, 0, stream>>>(ampn, topidx, pw, jobs);

    // conv inputs
    k_fill_imp<<<((NB * SMAXP * NCH) + 255) / 256, 256, 0, stream>>>(x, imp);
    k_cwt<<<((NB * NSCALES * T_TOT * NCH) + 255) / 256, 256, 0, stream>>>(x, dfilt, cwin, sp);

    // period branch: 3 x (conv 32->64 + gelu, conv 64->32)
    for (int k = 0; k < 3; ++k) {
        k_conv<<<dim3(SMAXP / 16, 4, NB), 32, 0, stream>>>(
            imp, SMAXP, 32, bswz1, bm1, 64, jobs, k, 1, mid, nullptr);
        k_conv<<<dim3(SMAXP / 16, 2, NB), 32, 0, stream>>>(
            mid, SMAXP, 64, bswz2, bm2, 32, jobs, k, 0, nullptr, pouts[k]);
    }
    // wavelet branch
    k_conv<<<dim3(SMAXW / 16, 4, NB), 32, 0, stream>>>(
        cwin, SMAXW, 32, bswz1, bm1, 64, jobs, 3, 1, mid, nullptr);
    k_conv<<<dim3(SMAXW / 16, 2, NB), 32, 0, stream>>>(
        mid, SMAXW, 64, bswz2, bm2, 32, jobs, 3, 0, nullptr, wout);

    // combine + residual
    k_combine<<<((NB * T_TOT * NCH) + 255) / 256, 256, 0, stream>>>(
        wout, pout0, pout1, pout2, pw, w_scale, b_scale, x, accb);

    // DCT gating block
    k_tr_btn_bnt<<<((NB * T_TOT * NCH) + 255) / 256, 256, 0, stream>>>(accb, xcl);
    k_dctmat<<<((T_TOT * T_TOT) + 255) / 256, 256, 0, stream>>>(Cm);
    k_gemm_f32<<<dim3((NB * NCH) / 16, T_TOT / 16), 32, 0, stream>>>(
        xcl, T_TOT, Cm, T_TOT, 0, nullptr, fbuf, NB * NCH, T_TOT, T_TOT, 0);
    k_rowstats<<<1, 256, 0, stream>>>(fbuf, stats);
    k_fcblock<<<((NB * T_TOT) + 255) / 256, 256, 0, stream>>>(
        fbuf, stats, xcl, ln_w, ln_b, w_fc1, w_fc2, x, out2);

    // final projection 672 -> 336
    k_tr_btn_bnt<<<((NB * T_TOT * NCH) + 255) / 256, 256, 0, stream>>>(out2, aproj);
    k_gemm_f32<<<dim3((NB * NCH) / 16, PRED / 16), 32, 0, stream>>>(
        aproj, T_TOT, w_proj, T_TOT, 1, b_proj, (float*)d_out, NB * NCH, PRED, T_TOT, 1);
}
